// NeuralHawkesIntentPredictor_59785944760758
// MI455X (gfx1250) — matrix operand order repaired and checked by reading, not compiled, not run on hardware
//
#include <hip/hip_runtime.h>
#include <hip/hip_bf16.h>
#include <math.h>

// ---------------------------------------------------------------------------
// Types
// ---------------------------------------------------------------------------
typedef __attribute__((ext_vector_type(16))) _Float16 v16h;
typedef __attribute__((ext_vector_type(8)))  float    v8f;

#define NB_ELEMS    256
#define NL_SEQ      200
#define NH_HID      768
#define ND_DIM      64
#define M_EV        51200          // B*L event rows
#define M_TOT       51456          // + B cls rows
#define NG          384            // 6*D gate cols
#define NSTG        6

// output layout (flat f32): probs[256*6] | tte[256] | cur_int[256*6] | int_hist[256*200*6]
#define OUT_PROBS   0
#define OUT_TTE     1536
#define OUT_CI      1792
#define OUT_IH      3328

// ---------------------------------------------------------------------------
// Math helpers (exact forms to match reference)
// ---------------------------------------------------------------------------
__device__ __forceinline__ float sigm_(float x)  { return 1.0f / (1.0f + expf(-x)); }
__device__ __forceinline__ float splus_(float x) { return (x > 20.0f) ? x : log1pf(expf(x)); }
__device__ __forceinline__ float gelu_(float x)  { return 0.5f * x * (1.0f + erff(x * 0.70710678118654752f)); }

// ---------------------------------------------------------------------------
// WMMA fragment loaders.
// A (16x32 f16, MxK): lane m=lane&15; half=lane>>4;
//   element j -> k = 16*half + j  (matches B packing).
// B (32x16 f16, KxN): lane n=lane&15, half=lane>>4; element j -> k = 16*half + j.
//   We pre-pack weights so a B fragment is one contiguous 32B load per lane.
// C/D (16x16 f32): VGPR j, lane: M = j + 8*(lane>>4), N = lane&15.
// ---------------------------------------------------------------------------
__device__ __forceinline__ v16h load_a_f32(const float* rowp, int half) {
  const float* q0 = rowp + half * 16;
  const float* q1 = rowp + half * 16 + 8;
  float4 f0 = *(const float4*)(q0);
  float4 f1 = *(const float4*)(q0 + 4);
  float4 f2 = *(const float4*)(q1);
  float4 f3 = *(const float4*)(q1 + 4);
  v16h a;
  a[0]=(_Float16)f0.x;  a[1]=(_Float16)f0.y;  a[2]=(_Float16)f0.z;  a[3]=(_Float16)f0.w;
  a[4]=(_Float16)f1.x;  a[5]=(_Float16)f1.y;  a[6]=(_Float16)f1.z;  a[7]=(_Float16)f1.w;
  a[8]=(_Float16)f2.x;  a[9]=(_Float16)f2.y;  a[10]=(_Float16)f2.z; a[11]=(_Float16)f2.w;
  a[12]=(_Float16)f3.x; a[13]=(_Float16)f3.y; a[14]=(_Float16)f3.z; a[15]=(_Float16)f3.w;
  return a;
}

__device__ __forceinline__ v16h load_a_h16(const _Float16* rowp, int half) {
  v16h a;
#pragma unroll
  for (int j = 0; j < 8; ++j) {
    a[j]     = rowp[half * 16 + j];
    a[8 + j] = rowp[half * 16 + 8 + j];
  }
  return a;
}

#define WMMA_F16(A, B, C) \
  __builtin_amdgcn_wmma_f32_16x16x32_f16(false, (A), false, (B), (short)0, (C), false, false)

// ---------------------------------------------------------------------------
// Pack kernels: weights -> f16 B-fragment layout.
// frag(kb,nb) occupies 512 halves: ((kb*NBfrag + nb)*32 + lane)*16 + j
// with lane = (k%32/16)*16 + (n%16), j = k%16.
// ---------------------------------------------------------------------------
__global__ void pack_wf_kernel(const float* __restrict__ We,  // (768,64)
                               const float* __restrict__ Wg,  // (128,384)
                               _Float16* __restrict__ Wf) {   // fused (768,384) fragments
  int idx = blockIdx.x * 256 + threadIdx.x;
  if (idx >= NH_HID * NG) return;
  int k = idx / NG, n = idx % NG;
  float s = 0.0f;
  for (int d = 0; d < ND_DIM; ++d) s += We[k * ND_DIM + d] * Wg[d * NG + n];
  int kb = k >> 5, kr = k & 31;
  int lane = (kr >> 4) * 16 + (n & 15), j = kr & 15, nb = n >> 4;
  Wf[((size_t)(kb * (NG / 16) + nb) * 32 + lane) * 16 + j] = (_Float16)s;
}

__global__ void pack_b_kernel(const float* __restrict__ W, _Float16* __restrict__ dst,
                              int K, int N, int Nfrag) {
  int idx = blockIdx.x * 256 + threadIdx.x;
  if (idx >= K * Nfrag) return;
  int k = idx / Nfrag, n = idx % Nfrag;
  float v = (n < N) ? W[k * N + n] : 0.0f;
  int NBf = Nfrag >> 4;
  int kb = k >> 5, kr = k & 31;
  int lane = (kr >> 4) * 16 + (n & 15), j = kr & 15, nb = n >> 4;
  dst[((size_t)(kb * NBf + nb) * 32 + lane) * 16 + j] = (_Float16)v;
}

// v0[n] = be@Wg_top + btime@Wg_bot + bg ;  v1[n] = Wtime@Wg_bot
__global__ void make_v_kernel(const float* __restrict__ be, const float* __restrict__ btime,
                              const float* __restrict__ Wtime, const float* __restrict__ Wg,
                              const float* __restrict__ bg,
                              float* __restrict__ v0, float* __restrict__ v1) {
  int n = blockIdx.x * 256 + threadIdx.x;
  if (n >= NG) return;
  float s0 = bg[n], s1 = 0.0f;
  for (int d = 0; d < ND_DIM; ++d) {
    s0 += be[d] * Wg[d * NG + n] + btime[d] * Wg[(ND_DIM + d) * NG + n];
    s1 += Wtime[d] * Wg[(ND_DIM + d) * NG + n];
  }
  v0[n] = s0; v1[n] = s1;
}

// ---------------------------------------------------------------------------
// K1: fused gate GEMM. rows [0,51200): event_embeds, [51200,51456): cls_embed.
// g[r,n] = act_n( X[r]@Wf[:,n] + v0[n] + days(r)*v1[n] ).
// 128x128 tile / WG; 8 waves as 4(M)x2(N), each wave 32Mx64N = 2x4 accumulators.
// ---------------------------------------------------------------------------
__global__ void __launch_bounds__(256)
gate_gemm_kernel(const float* __restrict__ ev, const float* __restrict__ cls,
                 const float* __restrict__ days, const _Float16* __restrict__ Wf,
                 const float* __restrict__ v0, const float* __restrict__ v1,
                 float* __restrict__ g) {
  const int w = threadIdx.x >> 5, lane = threadIdx.x & 31;
  const int half = lane >> 4, lr = lane & 15;
  const int mw = w & 3, nw = w >> 2;
  const int m0 = blockIdx.x * 128 + mw * 32;
  const int n0 = blockIdx.y * 128 + nw * 64;

  const int r0 = m0 + lr, r1 = r0 + 16;
  const float* pa0 = (r0 < M_EV) ? ev + (size_t)r0 * NH_HID : cls + (size_t)(r0 - M_EV) * NH_HID;
  const float* pa1 = (r1 < M_EV) ? ev + (size_t)r1 * NH_HID : cls + (size_t)(r1 - M_EV) * NH_HID;

  v8f acc[2][4] = {};
  for (int kk = 0; kk < NH_HID; kk += 32) {
    __builtin_prefetch(pa0 + kk + 64, 0, 1);
    __builtin_prefetch(pa1 + kk + 64, 0, 1);
    v16h a0 = load_a_f32(pa0 + kk, half);
    v16h a1 = load_a_f32(pa1 + kk, half);
    const int kb = kk >> 5;
#pragma unroll
    for (int ni = 0; ni < 4; ++ni) {
      const int nb = (n0 >> 4) + ni;
      v16h b = *(const v16h*)(Wf + ((size_t)(kb * (NG / 16) + nb) * 32 + lane) * 16);
      acc[0][ni] = WMMA_F16(a0, b, acc[0][ni]);
      acc[1][ni] = WMMA_F16(a1, b, acc[1][ni]);
    }
  }

#pragma unroll
  for (int mi = 0; mi < 2; ++mi) {
#pragma unroll
    for (int j = 0; j < 8; ++j) {
      const int r = m0 + mi * 16 + j + 8 * half;
      const float dy = (r < M_EV) ? days[r] : 0.0f;   // cls row == dt=0 path
#pragma unroll
      for (int ni = 0; ni < 4; ++ni) {
        const int n = n0 + ni * 16 + lr;
        float x = acc[mi][ni][j] + v0[n] + dy * v1[n];
        const int gsel = n >> 6;                       // 0..5 : i f z o dl cb
        float y;
        if (gsel == 2)      y = tanhf(x);
        else if (gsel == 4) y = splus_(x);
        else if (gsel == 5) y = x;
        else                y = sigm_(x);
        g[(size_t)r * NG + n] = y;
      }
    }
  }
}

// ---------------------------------------------------------------------------
// K2: CT-LSTM scan (serial in L) + final dt=0 update. One block per batch row.
// ---------------------------------------------------------------------------
__global__ void __launch_bounds__(64)
scan_kernel(const float* __restrict__ g, const float* __restrict__ days,
            float* __restrict__ h) {
  const int b = blockIdx.x, d = threadIdx.x;
  float c = 0.0f;
  const float* gb = g + (size_t)b * NL_SEQ * NG;
  for (int l = 0; l < NL_SEQ; ++l) {
    const float* gr = gb + (size_t)l * NG;
    float i  = gr[d],        f  = gr[64 + d],  z  = gr[128 + d];
    float o  = gr[192 + d],  dl = gr[256 + d], cb = gr[320 + d];
    float dt = fmaxf(days[b * NL_SEQ + l], 0.0f);
    c = f * c + i * z;
    float cdec = cb + (c - cb) * expf(-dl * dt);
    h[((size_t)b * NL_SEQ + l) * ND_DIM + d] = o * tanhf(cdec);
  }
  // final update at dt=0: c_dec == c
  const float* gr = g + (size_t)(M_EV + b) * NG;
  float c2 = gr[64 + d] * c + gr[d] * gr[128 + d];
  h[((size_t)(M_EV + b)) * ND_DIM + d] = gr[192 + d] * tanhf(c2);
}

// ---------------------------------------------------------------------------
// K3: intensity head over all 51456 h-rows: softplus(gelu(h@Wi1+bi1)@Wi2+bi2).
// Two WMMA stages with an LDS re-fragmentation in between. One 16-row tile/wave.
// ---------------------------------------------------------------------------
__global__ void __launch_bounds__(256)
intensity_kernel(const float* __restrict__ h, const _Float16* __restrict__ Wi1f,
                 const float* __restrict__ bi1, const _Float16* __restrict__ Wi2f,
                 const float* __restrict__ bi2, float* __restrict__ out) {
  __shared__ _Float16 lds[8 * 512];
  const int w = threadIdx.x >> 5, lane = threadIdx.x & 31;
  const int half = lane >> 4, lr = lane & 15;
  const int mt = blockIdx.x * 8 + w;
  const float* rowp = h + ((size_t)mt * 16 + lr) * ND_DIM;

  v8f acc[2] = {};
  for (int kk = 0; kk < ND_DIM; kk += 32) {
    v16h a = load_a_f32(rowp + kk, half);
#pragma unroll
    for (int ni = 0; ni < 2; ++ni) {
      v16h b = *(const v16h*)(Wi1f + ((size_t)((kk >> 5) * 2 + ni) * 32 + lane) * 16);
      acc[ni] = WMMA_F16(a, b, acc[ni]);
    }
  }

  _Float16* myl = lds + w * 512;                       // 16x32 row-major tile
#pragma unroll
  for (int ni = 0; ni < 2; ++ni)
#pragma unroll
    for (int j = 0; j < 8; ++j) {
      const int m = j + 8 * half, col = ni * 16 + lr;
      myl[m * 32 + col] = (_Float16)gelu_(acc[ni][j] + bi1[col]);
    }
  __syncthreads();

  v16h a2 = load_a_h16(myl + lr * 32, half);
  v8f acc2 = {};
  v16h b2 = *(const v16h*)(Wi2f + (size_t)lane * 16);
  acc2 = WMMA_F16(a2, b2, acc2);

  if (lr < NSTG) {
#pragma unroll
    for (int j = 0; j < 8; ++j) {
      const int r = mt * 16 + j + 8 * half;
      float y = splus_(acc2[j] + bi2[lr]);
      if (r < M_EV) out[OUT_IH + (size_t)r * NSTG + lr] = y;           // (b*L+l)*6
      else          out[OUT_CI + (size_t)(r - M_EV) * NSTG + lr] = y;  // current
    }
  }
}

// ---------------------------------------------------------------------------
// K4: direct logits: gelu(cls@Ws1+bs1)@Ws2+bs2 -> (256,6). 16 rows per block.
// ---------------------------------------------------------------------------
__global__ void __launch_bounds__(256)
direct_kernel(const float* __restrict__ cls, const _Float16* __restrict__ Ws1f,
              const float* __restrict__ bs1, const _Float16* __restrict__ Ws2f,
              const float* __restrict__ bs2, float* __restrict__ dlog) {
  __shared__ _Float16 lds[16 * 384];
  const int w = threadIdx.x >> 5, lane = threadIdx.x & 31;
  const int half = lane >> 4, lr = lane & 15;
  const int mt = blockIdx.x;
  const float* rowp = cls + ((size_t)mt * 16 + lr) * NH_HID;

  v8f acc[3] = {};
  for (int kk = 0; kk < NH_HID; kk += 32) {
    v16h a = load_a_f32(rowp + kk, half);
#pragma unroll
    for (int ni = 0; ni < 3; ++ni) {
      const int nb = w * 3 + ni;
      v16h b = *(const v16h*)(Ws1f + ((size_t)((kk >> 5) * (NG / 16) + nb) * 32 + lane) * 16);
      acc[ni] = WMMA_F16(a, b, acc[ni]);
    }
  }
#pragma unroll
  for (int ni = 0; ni < 3; ++ni)
#pragma unroll
    for (int j = 0; j < 8; ++j) {
      const int m = j + 8 * half, col = w * 48 + ni * 16 + lr;
      lds[m * 384 + col] = (_Float16)gelu_(acc[ni][j] + bs1[col]);
    }
  __syncthreads();

  if (w == 0) {                                  // whole wave active: EXEC all-1s
    v8f acc2 = {};
    for (int kk = 0; kk < 384; kk += 32) {
      v16h a2 = load_a_h16(lds + lr * 384 + kk, half);
      v16h b2 = *(const v16h*)(Ws2f + ((size_t)(kk >> 5) * 32 + lane) * 16);
      acc2 = WMMA_F16(a2, b2, acc2);
    }
    if (lr < NSTG)
#pragma unroll
      for (int j = 0; j < 8; ++j) {
        const int r = mt * 16 + j + 8 * half;
        dlog[r * NSTG + lr] = acc2[j] + bs2[lr];
      }
  }
}

// ---------------------------------------------------------------------------
// K5: tail — softmax(direct + log(cur_int)) and TTE head. One thread per batch.
// ---------------------------------------------------------------------------
__global__ void __launch_bounds__(256)
tail_kernel(const float* __restrict__ h, const float* __restrict__ Wq1,
            const float* __restrict__ bq1, const float* __restrict__ Wq2,
            const float* __restrict__ bq2, const float* __restrict__ dlog,
            float* __restrict__ out) {
  const int b = blockIdx.x * blockDim.x + threadIdx.x;
  if (b >= NB_ELEMS) return;

  float lg[NSTG], mx = -3.0e38f;
#pragma unroll
  for (int k = 0; k < NSTG; ++k) {
    lg[k] = dlog[b * NSTG + k] + logf(out[OUT_CI + b * NSTG + k]);
    mx = fmaxf(mx, lg[k]);
  }
  float s = 0.0f;
#pragma unroll
  for (int k = 0; k < NSTG; ++k) { lg[k] = expf(lg[k] - mx); s += lg[k]; }
#pragma unroll
  for (int k = 0; k < NSTG; ++k) out[OUT_PROBS + b * NSTG + k] = lg[k] / s;

  const float* hb = h + (size_t)(M_EV + b) * ND_DIM;
  float t = bq2[0];
  for (int j = 0; j < 32; ++j) {
    float a = bq1[j];
    for (int d = 0; d < ND_DIM; ++d) a += hb[d] * Wq1[d * 32 + j];
    t += gelu_(a) * Wq2[j];
  }
  out[OUT_TTE + b] = splus_(t);
}

// ---------------------------------------------------------------------------
// Launch
// ---------------------------------------------------------------------------
extern "C" void kernel_launch(void* const* d_in, const int* in_sizes, int n_in,
                              void* d_out, int out_size, void* d_ws, size_t ws_size,
                              hipStream_t stream) {
  const float* cls   = (const float*)d_in[0];
  const float* ev    = (const float*)d_in[1];
  const float* days  = (const float*)d_in[2];
  const float* We    = (const float*)d_in[3];
  const float* be    = (const float*)d_in[4];
  const float* Wtime = (const float*)d_in[5];
  const float* btime = (const float*)d_in[6];
  const float* Wg    = (const float*)d_in[7];
  const float* bg    = (const float*)d_in[8];
  const float* Wi1   = (const float*)d_in[9];
  const float* bi1   = (const float*)d_in[10];
  const float* Wi2   = (const float*)d_in[11];
  const float* bi2   = (const float*)d_in[12];
  const float* Ws1   = (const float*)d_in[13];
  const float* bs1   = (const float*)d_in[14];
  const float* Ws2   = (const float*)d_in[15];
  const float* bs2   = (const float*)d_in[16];
  const float* Wq1   = (const float*)d_in[17];
  const float* bq1   = (const float*)d_in[18];
  const float* Wq2   = (const float*)d_in[19];
  const float* bq2   = (const float*)d_in[20];
  float* out = (float*)d_out;

  // workspace layout (all offsets 64B aligned)
  char* ws = (char*)d_ws;
  float*     ws_g    = (float*)(ws + 0);                         // 51456*384 f32
  float*     ws_h    = (float*)(ws + 79036416);                  // 51456*64  f32
  _Float16*  ws_Wf   = (_Float16*)(ws + 92209152);               // 768*384 f16 frags
  _Float16*  ws_Ws1f = (_Float16*)(ws + 92798976);               // 768*384 f16 frags
  float*     ws_v0   = (float*)(ws + 93388800);                  // 384
  float*     ws_v1   = (float*)(ws + 93390336);                  // 384
  _Float16*  ws_Wi1f = (_Float16*)(ws + 93391872);               // 64*32 f16 frags
  _Float16*  ws_Wi2f = (_Float16*)(ws + 93395968);               // 32*16 f16 frags (padded)
  _Float16*  ws_Ws2f = (_Float16*)(ws + 93396992);               // 384*16 f16 frags (padded)
  float*     ws_dlog = (float*)(ws + 93409280);                  // 256*6

  // --- weight packing (tiny, L2-resident afterwards) ---
  pack_wf_kernel<<<(NH_HID * NG + 255) / 256, 256, 0, stream>>>(We, Wg, ws_Wf);
  pack_b_kernel<<<(NH_HID * NG + 255) / 256, 256, 0, stream>>>(Ws1, ws_Ws1f, NH_HID, NG, NG);
  pack_b_kernel<<<(64 * 32 + 255) / 256, 256, 0, stream>>>(Wi1, ws_Wi1f, 64, 32, 32);
  pack_b_kernel<<<(32 * 16 + 255) / 256, 256, 0, stream>>>(Wi2, ws_Wi2f, 32, NSTG, 16);
  pack_b_kernel<<<(384 * 16 + 255) / 256, 256, 0, stream>>>(Ws2, ws_Ws2f, 384, NSTG, 16);
  make_v_kernel<<<2, 256, 0, stream>>>(be, btime, Wtime, Wg, bg, ws_v0, ws_v1);

  // --- main fused gate GEMM + activation epilogue (HBM-bound pass) ---
  gate_gemm_kernel<<<dim3(M_TOT / 128, NG / 128), 256, 0, stream>>>(
      ev, cls, days, ws_Wf, ws_v0, ws_v1, ws_g);

  // --- CT-LSTM scan + final update ---
  scan_kernel<<<NB_ELEMS, 64, 0, stream>>>(ws_g, days, ws_h);

  // --- intensity head (history + current) ---
  intensity_kernel<<<M_TOT / 128, 256, 0, stream>>>(ws_h, ws_Wi1f, bi1, ws_Wi2f, bi2, out);

  // --- direct logits ---
  direct_kernel<<<NB_ELEMS / 16, 256, 0, stream>>>(cls, ws_Ws1f, bs1, ws_Ws2f, bs2, ws_dlog);

  // --- softmax + tte ---
  tail_kernel<<<1, 256, 0, stream>>>(ws_h, Wq1, bq1, Wq2, bq2, ws_dlog, out);
}
